// MultiheadAttention_60129542924
// MI455X (gfx1250) — matrix operand order repaired and checked by reading, not compiled, not run on hardware
//
#include <hip/hip_runtime.h>

// Problem sizes (fixed by the reference)
#define EMBED    1024
#define HEADS    16
#define HEAD_DIM 64
#define BSZ      32
#define TGT      512
#define SRC      512
#define BH       512   // BSZ*HEADS

typedef float    v8f  __attribute__((ext_vector_type(8)));
typedef _Float16 v8h  __attribute__((ext_vector_type(8)));
typedef _Float16 v16h __attribute__((ext_vector_type(16)));

// D(16x16 f32) = A(16x32 f16) * B(32x16 f16) + C   — codegen-confirmed builtin
__device__ __forceinline__ v8f wmma_f16(v16h a, v16h b, v8f c) {
  return __builtin_amdgcn_wmma_f32_16x16x32_f16(
      false, a, false, b, (short)0, c, false, false);
}

__device__ __forceinline__ v16h cat16(v8h lo, v8h hi) {
  return __builtin_shufflevector(lo, hi, 0, 1, 2, 3, 4, 5, 6, 7,
                                 8, 9, 10, 11, 12, 13, 14, 15);
}

// f32 -> f16 elementwise (weights; 1M elements, multiple of 8)
__global__ __launch_bounds__(256) void cvt_f32_f16(
    const float* __restrict__ in, _Float16* __restrict__ out, int n) {
  const int i = (blockIdx.x * 256 + threadIdx.x) * 8;
  if (i + 7 < n) {
    const float4 a = *(const float4*)(in + i);
    const float4 b = *(const float4*)(in + i + 4);
    v8h h = {(_Float16)a.x, (_Float16)a.y, (_Float16)a.z, (_Float16)a.w,
             (_Float16)b.x, (_Float16)b.y, (_Float16)b.z, (_Float16)b.w};
    *(v8h*)(out + i) = h;
  }
}

// ---------------------------------------------------------------------------
// Projection GEMM:  acc_f32[m,n] = sum_k X(m,k) * W16[n,k] + bias[n]
//   m = t*BSZ + b; X(m,k) at X + t*strideT + b*strideB + k (elements)
//   xIsF16: X is fp16 (ao) vs fp32 (raw inputs; converted while staging)
//   mode: 0 = f32 row store, 1 = f16 row store, 2 = f16 transposed store
//         (mode 2 writes V as [t][j=0..63][s=b*16+h] for K-contiguous B reads)
// Block: 256 threads = 8 waves; block tile 16(M) x 128(N); fp16 A tile in LDS.
// ---------------------------------------------------------------------------
__global__ __launch_bounds__(256) void proj_gemm16(
    const void* __restrict__ Xv, const _Float16* __restrict__ W16,
    const float* __restrict__ bias, void* __restrict__ Yv,
    long strideT, long strideB, int xIsF16, int mode) {
  extern __shared__ _Float16 lA[];          // 16 x 1024 halves = 32 KB
  const int tid  = threadIdx.x;
  const int wave = tid >> 5;
  const int lane = tid & 31;
  const int m0   = blockIdx.x * 16;
  const int n0   = blockIdx.y * 128 + wave * 16;

  // Stage A tile (convert f32->f16 on the fly if needed)
  for (int i = tid; i < 16 * (EMBED / 8); i += 256) {
    const int r  = i >> 7;                  // / 128
    const int c8 = (i & 127) * 8;
    const int m  = m0 + r;
    const size_t off = (size_t)(m >> 5) * strideT + (size_t)(m & 31) * strideB + c8;
    v8h h;
    if (xIsF16) {
      h = *(const v8h*)((const _Float16*)Xv + off);
    } else {
      const float* xp = (const float*)Xv + off;
      const float4 a = *(const float4*)xp;
      const float4 b = *(const float4*)(xp + 4);
      h = (v8h){(_Float16)a.x, (_Float16)a.y, (_Float16)a.z, (_Float16)a.w,
                (_Float16)b.x, (_Float16)b.y, (_Float16)b.z, (_Float16)b.w};
    }
    *(v8h*)(lA + r * EMBED + c8) = h;
  }
  __syncthreads();

  const int lrow  = lane & 15;
  const int kaoff = (lane >> 4) * 16;       // A: 16 contiguous K
  const int kboff = (lane >> 4) * 16;       // B: 16 contiguous K
  const int mhi   = (lane >> 4) * 8;        // C/D: high lanes hold M+8

  const _Float16* arow = lA + lrow * EMBED;
  const _Float16* wrow = W16 + (size_t)(n0 + lrow) * EMBED;
  v8f c = {0.f, 0.f, 0.f, 0.f, 0.f, 0.f, 0.f, 0.f};
  for (int k0 = 0; k0 < EMBED; k0 += 32) {
    const v8h alo = *(const v8h*)(arow + k0 + kaoff);
    const v8h ahi = *(const v8h*)(arow + k0 + kaoff + 8);
    const v8h blo = *(const v8h*)(wrow + k0 + kboff);
    const v8h bhi = *(const v8h*)(wrow + k0 + kboff + 8);
    c = wmma_f16(cat16(alo, ahi), cat16(blo, bhi), c);
  }

  const int   nn = n0 + lrow;
  const float bn = bias[nn];
  if (mode == 0) {                          // f32 rows (final output)
    float* Y = (float*)Yv;
#pragma unroll
    for (int r = 0; r < 8; ++r)
      Y[(size_t)(m0 + r + mhi) * EMBED + nn] = c[r] + bn;
  } else if (mode == 1) {                   // f16 rows (q, k)
    _Float16* Y = (_Float16*)Yv;
#pragma unroll
    for (int r = 0; r < 8; ++r)
      Y[(size_t)(m0 + r + mhi) * EMBED + nn] = (_Float16)(c[r] + bn);
  } else {                                  // f16 transposed per-t (v)
    _Float16* Y = (_Float16*)Yv;
    const int j  = nn & 63;
    const int hh = nn >> 6;
#pragma unroll
    for (int r = 0; r < 8; ++r) {
      const int m = m0 + r + mhi;
      Y[(size_t)(m >> 5) * (64 * 512) + j * 512 + (m & 31) * 16 + hh] =
          (_Float16)(c[r] + bn);
    }
  }
}

// ---------------------------------------------------------------------------
// Fused attention, one workgroup per t:
//   scores = q_t @ k_t^T / 8 (f16 x f16 -> f32) ; softmax (f32) ;
//   ao_t = attn(f16) @ v_t(f16, pre-transposed) ; need_weights h-mean in LDS
// LDS: 64x512 f32 scores (128K) + 64x512 f16 attn (64K) + 32x512 f32 w (64K)
// ---------------------------------------------------------------------------
__global__ __launch_bounds__(256) void attn_fused16(
    const _Float16* __restrict__ qp, const _Float16* __restrict__ kp,
    const _Float16* __restrict__ vT, _Float16* __restrict__ ao,
    float* __restrict__ wout) {
  extern __shared__ char smem[];
  float*    sc   = (float*)smem;                          // [64][512] f32
  _Float16* sch  = (_Float16*)(smem + 64 * 512 * 4);      // [64][512] f16
  float*    wacc = (float*)(smem + 64 * 512 * 4 + 64 * 512 * 2); // [32][512]

  const int t    = blockIdx.x;
  const int tid  = threadIdx.x;
  const int wave = tid >> 5;
  const int lane = tid & 31;
  const int lrow  = lane & 15;
  const int kaoff = (lane >> 4) * 16;
  const int kboff = (lane >> 4) * 16;
  const int mhi   = (lane >> 4) * 8;

  const _Float16* q = qp + (size_t)t * BH * HEAD_DIM;   // [512][64]
  const _Float16* k = kp + (size_t)t * BH * HEAD_DIM;   // [512][64]
  const _Float16* v = vT + (size_t)t * HEAD_DIM * BH;   // [64][512] transposed

  for (int i = tid; i < 32 * 512; i += 256) wacc[i] = 0.f;
  __syncthreads();

  for (int qb = 0; qb < 8; ++qb) {          // 64 query rows per pass
    const int r0 = qb * 64;

    // ---- scores: 4 m-tiles x 32 n-tiles = 128 tiles, K=64 -> 2 WMMAs ----
    for (int tt = wave; tt < 128; tt += 8) {
      const int mt = tt >> 5;
      const int nt = tt & 31;
      const _Float16* arow = q + (size_t)(r0 + mt * 16 + lrow) * HEAD_DIM;
      const _Float16* brow = k + (size_t)(nt * 16 + lrow) * HEAD_DIM;
      v8f c = {0.f, 0.f, 0.f, 0.f, 0.f, 0.f, 0.f, 0.f};
#pragma unroll
      for (int k0 = 0; k0 < HEAD_DIM; k0 += 32) {
        const v8h alo = *(const v8h*)(arow + k0 + kaoff);
        const v8h ahi = *(const v8h*)(arow + k0 + kaoff + 8);
        const v8h blo = *(const v8h*)(brow + k0 + kboff);
        const v8h bhi = *(const v8h*)(brow + k0 + kboff + 8);
        c = wmma_f16(cat16(alo, ahi), cat16(blo, bhi), c);
      }
#pragma unroll
      for (int r = 0; r < 8; ++r)
        sc[(mt * 16 + r + mhi) * 512 + nt * 16 + lrow] = c[r] * 0.125f;
    }
    __syncthreads();

    // ---- softmax over 512 columns of each of the 64 rows (f32) ----
    if (tid < 64) {
      float* row = sc + tid * 512;
      float mx = row[0];
      for (int s = 1; s < 512; ++s) mx = fmaxf(mx, row[s]);
      float sum = 0.f;
      for (int s = 0; s < 512; ++s) {
        const float e = __expf(row[s] - mx);
        row[s] = e;
        sum += e;
      }
      const float inv = 1.f / sum;
      for (int s = 0; s < 512; ++s) row[s] *= inv;
    }
    __syncthreads();

    // ---- f16 copy of attn + need_weights accumulation ----
    for (int i = tid; i < 64 * 512; i += 256) sch[i] = (_Float16)sc[i];
    for (int i = tid; i < 32 * 512; i += 256)
      wacc[i] += sc[i] + sc[i + 32 * 512];  // rows bv and bv+32 share bh%32
    __syncthreads();

    // ---- context: attn(64x512) @ v(512x64); 16 tiles, K=512 -> 16 WMMAs ----
    for (int tt = wave; tt < 16; tt += 8) {
      const int mt = tt >> 2;
      const int nt = tt & 3;
      const int j0 = nt * 16 + lrow;
      const _Float16* arow = sch + (mt * 16 + lrow) * 512;
      const _Float16* brow = v + (size_t)j0 * 512;   // K-contiguous (vT)
      v8f c = {0.f, 0.f, 0.f, 0.f, 0.f, 0.f, 0.f, 0.f};
      for (int k0 = 0; k0 < 512; k0 += 32) {
        const v8h alo = *(const v8h*)(arow + k0 + kaoff);
        const v8h ahi = *(const v8h*)(arow + k0 + kaoff + 8);
        const v8h blo = *(const v8h*)(brow + k0 + kboff);
        const v8h bhi = *(const v8h*)(brow + k0 + kboff + 8);
        c = wmma_f16(cat16(alo, ahi), cat16(blo, bhi), c);
      }
#pragma unroll
      for (int r = 0; r < 8; ++r) {
        const int m = r0 + mt * 16 + r + mhi;
        ao[((size_t)t * BH + m) * HEAD_DIM + j0] = (_Float16)c[r];
      }
    }
    __syncthreads();                        // sc/sch reused next qb pass
  }

  // wout is [B=32, T, S]; mean over the 16 "view heads"
  for (int i = tid; i < 32 * 512; i += 256) {
    const int bv = i >> 9;
    const int s  = i & 511;
    wout[((size_t)bv * TGT + t) * SRC + s] = wacc[i] * (1.0f / 16.0f);
  }
}

// ---------------------------------------------------------------------------
extern "C" void kernel_launch(void* const* d_in, const int* in_sizes, int n_in,
                              void* d_out, int out_size, void* d_ws,
                              size_t ws_size, hipStream_t stream) {
  const float* query = (const float*)d_in[0];   // [32,512,1024]
  const float* key   = (const float*)d_in[1];   // [512,32,1024]
  const float* value = (const float*)d_in[2];   // [512,32,1024]
  const float* Wq = (const float*)d_in[3];  const float* bq = (const float*)d_in[4];
  const float* Wk = (const float*)d_in[5];  const float* bk = (const float*)d_in[6];
  const float* Wv = (const float*)d_in[7];  const float* bv = (const float*)d_in[8];
  const float* Wo = (const float*)d_in[9];  const float* bo = (const float*)d_in[10];

  float* out = (float*)d_out;                          // [T,B,D] f32
  float* wts = out + (size_t)TGT * BSZ * EMBED;        // [B,T,S] f32

  // Workspace (all fp16): 4 weight copies + q, k, vT, ao  (~143 MB)
  const size_t WSZ = (size_t)EMBED * EMBED;            // 1,048,576
  const size_t HSZ = (size_t)TGT * BH * HEAD_DIM;      // 16,777,216
  _Float16* ws   = (_Float16*)d_ws;
  _Float16* Wq16 = ws + 0 * WSZ;
  _Float16* Wk16 = ws + 1 * WSZ;
  _Float16* Wv16 = ws + 2 * WSZ;
  _Float16* Wo16 = ws + 3 * WSZ;
  _Float16* qp   = ws + 4 * WSZ;
  _Float16* kp   = qp + HSZ;
  _Float16* vT   = kp + HSZ;
  _Float16* ao   = vT + HSZ;

  const size_t ldsProj = (size_t)16 * EMBED * sizeof(_Float16);       // 32 KB
  const size_t ldsAttn = (size_t)(64 * 512 * 4 + 64 * 512 * 2 + 32 * 512 * 4); // 256 KB
  (void)hipFuncSetAttribute((const void*)proj_gemm16,
                            hipFuncAttributeMaxDynamicSharedMemorySize,
                            (int)ldsProj);
  (void)hipFuncSetAttribute((const void*)attn_fused16,
                            hipFuncAttributeMaxDynamicSharedMemorySize,
                            (int)ldsAttn);

  // Convert the four weight matrices to fp16 (L2-resident thereafter)
  const dim3 cg(WSZ / (256 * 8));
  cvt_f32_f16<<<cg, 256, 0, stream>>>(Wq, Wq16, (int)WSZ);
  cvt_f32_f16<<<cg, 256, 0, stream>>>(Wk, Wk16, (int)WSZ);
  cvt_f32_f16<<<cg, 256, 0, stream>>>(Wv, Wv16, (int)WSZ);
  cvt_f32_f16<<<cg, 256, 0, stream>>>(Wo, Wo16, (int)WSZ);

  const dim3 gg(TGT * BSZ / 16, EMBED / 128);   // 1024 x 8 blocks
  const dim3 bb(256);

  // Q gathers through the [bsz,tgt,d] -> [tgt,bsz,d] transpose via strides
  proj_gemm16<<<gg, bb, ldsProj, stream>>>(query, Wq16, bq, qp,
                                           (long)EMBED, (long)TGT * EMBED, 0, 1);
  proj_gemm16<<<gg, bb, ldsProj, stream>>>(key, Wk16, bk, kp,
                                           (long)BSZ * EMBED, (long)EMBED, 0, 1);
  proj_gemm16<<<gg, bb, ldsProj, stream>>>(value, Wv16, bv, vT,
                                           (long)BSZ * EMBED, (long)EMBED, 0, 2);

  attn_fused16<<<dim3(TGT), bb, ldsAttn, stream>>>(qp, kp, vT, ao, wts);

  // Output projection: ao is [T,B,D]-contiguous fp16; result f32 + bias
  proj_gemm16<<<gg, bb, ldsProj, stream>>>(ao, Wo16, bo, out,
                                           (long)BSZ * EMBED, (long)EMBED, 1, 0);
}
